// GraphNetwork_20985210209012
// MI455X (gfx1250) — compile-verified, matrix operand order checked by reading, not run on hardware
//
#include <hip/hip_runtime.h>
#include <hip/hip_bf16.h>
#include <math.h>

#define N_NODES   50000
#define N_EDGES   800000
#define D_IN      8
#define D_MODEL   128
#define N_CLASSES 2

typedef __attribute__((ext_vector_type(16))) _Float16 v16h;
typedef __attribute__((ext_vector_type(8)))  _Float16 v8h;
typedef __attribute__((ext_vector_type(8)))  float    v8f;

union V16H { v16h v; v8h h[2]; };

__device__ inline v8f wmma_f16(v16h a, v16h b, v8f c) {
  // D = A(16x32 f16) * B(32x16 f16) + C(16x16 f32)
  return __builtin_amdgcn_wmma_f32_16x16x32_f16(
      /*neg_a=*/false, a, /*neg_b=*/false, b,
      /*c_mod=*/(short)0, c, /*reuse_a=*/false, /*reuse_b=*/false);
}

__device__ inline v8h cvt8(const float* __restrict__ p) {
  v8h r;
#pragma unroll
  for (int i = 0; i < 8; ++i) r[i] = (_Float16)p[i];
  return r;
}

__device__ inline v8h cvt8_fma(const float* __restrict__ p,
                               const float* __restrict__ q, float s) {
  v8h r;
#pragma unroll
  for (int i = 0; i < 8; ++i) r[i] = (_Float16)(p[i] + q[i] * s);
  return r;
}

// 16-byte global -> LDS async DMA (per-lane), tracked by ASYNCcnt.
__device__ inline void async_b128(unsigned int lds_off, const void* g) {
  unsigned long long ga = (unsigned long long)(uintptr_t)g;
  asm volatile("global_load_async_to_lds_b128 %0, %1, off"
               :: "v"(lds_off), "v"(ga) : "memory");
}

__device__ inline void wait_async0() {
#if defined(__has_builtin) && __has_builtin(__builtin_amdgcn_s_wait_asynccnt)
  __builtin_amdgcn_s_wait_asynccnt(0);
#else
  asm volatile("s_wait_asynccnt 0" ::: "memory");
#endif
}

// ---------- degree count: deg[dst] += 1 ----------
__global__ void k_deg(const int* __restrict__ dst, float* __restrict__ deg) {
  int e = blockIdx.x * blockDim.x + threadIdx.x;
  if (e < N_EDGES) atomicAdd(&deg[dst[e]], 1.0f);
}

// ---------- dinv = rsqrt(deg + 1) (in place) ----------
__global__ void k_dinv(float* __restrict__ deg_dinv) {
  int i = blockIdx.x * blockDim.x + threadIdx.x;
  if (i < N_NODES) deg_dinv[i] = rsqrtf(deg_dinv[i] + 1.0f);
}

// ---------- one-shot: transposed f16 weight images for async LDS staging ----------
__global__ void k_cvtW(const float* __restrict__ W2, const float* __restrict__ Wlin1,
                       _Float16* __restrict__ W2T, _Float16* __restrict__ WlinT) {
  int t = blockIdx.x * blockDim.x + threadIdx.x;
  if (t < D_MODEL * D_MODEL) {            // W2^T: [col][k] (128x128)
    int col = t >> 7, k = t & 127;
    W2T[col * D_MODEL + k] = (_Float16)W2[k * D_MODEL + col];
  }
  if (t < 2 * D_MODEL * D_MODEL) {        // Wlin1^T: [col][k] (128x256)
    int col = t >> 8, k = t & 255;
    WlinT[col * 256 + k] = (_Float16)Wlin1[k * D_MODEL + col];
  }
}

// ---------- layer-1 aggregation on RAW x (8 floats/edge, W1 commuted out) ----------
__global__ void k_xagg(const int* __restrict__ src, const int* __restrict__ dst,
                       const float* __restrict__ dinv, const float* __restrict__ x,
                       float* __restrict__ xagg) {
  int e = blockIdx.x * blockDim.x + threadIdx.x;
  if (e >= N_EDGES) return;
  int s = src[e], d = dst[e];
  float norm = dinv[s] * dinv[d];
  const float4* xs = (const float4*)(x + s * D_IN);
  float4 a = xs[0], b = xs[1];
  float* o = xagg + d * D_IN;
  atomicAdd(o + 0, a.x * norm); atomicAdd(o + 1, a.y * norm);
  atomicAdd(o + 2, a.z * norm); atomicAdd(o + 3, a.w * norm);
  atomicAdd(o + 4, b.x * norm); atomicAdd(o + 5, b.y * norm);
  atomicAdd(o + 6, b.z * norm); atomicAdd(o + 7, b.w * norm);
}

// ---------- layer 1: h1 = relu((xagg + x*dinv^2) @ W1 + b1)  (K=8, VALU is fine) ----------
__global__ void k_layer1(const float* __restrict__ x, const float* __restrict__ xagg,
                         const float* __restrict__ dinv, const float* __restrict__ W1,
                         const float* __restrict__ b1, float* __restrict__ h1) {
  int t = blockIdx.x * blockDim.x + threadIdx.x;
  if (t >= N_NODES * D_MODEL) return;
  int n = t >> 7, o = t & 127;
  float dv = dinv[n];
  float dv2 = dv * dv;
  float acc = b1[o];
#pragma unroll
  for (int k = 0; k < D_IN; ++k)
    acc += (xagg[n * D_IN + k] + x[n * D_IN + k] * dv2) * W1[k * D_MODEL + o];
  h1[t] = fmaxf(acc, 0.0f);
}

// ---------- layer-2 aggregation: P[dst] += h1[src]*norm  (1 wave = 1 edge, 4 floats/lane) ----------
__global__ void k_hagg(const int* __restrict__ src, const int* __restrict__ dst,
                       const float* __restrict__ dinv, const float* __restrict__ h1,
                       float* __restrict__ P) {
  int t = blockIdx.x * blockDim.x + threadIdx.x;
  int e = t >> 5;
  int lane = t & 31;
  if (e >= N_EDGES) return;
  int s = src[e], d = dst[e];
  float norm = dinv[s] * dinv[d];
  int k = lane * 4;
  float4 hv = *(const float4*)(h1 + s * D_MODEL + k);
  float* o = P + d * D_MODEL + k;
  atomicAdd(o + 0, hv.x * norm); atomicAdd(o + 1, hv.y * norm);
  atomicAdd(o + 2, hv.z * norm); atomicAdd(o + 3, hv.w * norm);
}

// ---------- layer 2 GEMM (WMMA): h2 = relu((P + h1*dinv^2) @ W2 + b2) ----------
// Wave computes a 16x128 tile (8 N-tiles of 16); W2^T (f16) DMA'd into LDS via async-to-LDS.
__global__ void __launch_bounds__(256) k_node_gemm(
    const float* __restrict__ P, const float* __restrict__ h1,
    const float* __restrict__ dinv, const _Float16* __restrict__ W2T,
    const float* __restrict__ b2, float* __restrict__ h2) {
  __shared__ _Float16 wT[D_MODEL * D_MODEL];  // W2^T: [col][k], 32 KB
  int tid = threadIdx.x;
  {
    unsigned int lds_base = (unsigned int)(uintptr_t)(void*)wT;  // addr[31:0] == LDS offset
#pragma unroll
    for (int i = 0; i < 8; ++i) {           // 2048 x 16B chunks / 256 threads
      int idx = (tid + i * 256) * 8;        // in halves
      async_b128(lds_base + (unsigned int)idx * 2u, W2T + idx);
    }
    wait_async0();
  }
  __syncthreads();

  int wave = tid >> 5, lane = tid & 31;
  int tile = blockIdx.x * 8 + wave;
  if (tile >= N_NODES / 16) return;  // wave-uniform: EXEC stays all-ones for WMMA

  int half = lane >> 4, mn = lane & 15;
  int node = tile * 16 + mn;
  float dv = dinv[node], dv2 = dv * dv;
  const float* Prow = P + node * D_MODEL;
  const float* Hrow = h1 + node * D_MODEL;

  V16H afr[4];  // K = 128 -> 4 k-steps of 32
#pragma unroll
  for (int ks = 0; ks < 4; ++ks) {
    int base = ks * 32 + half * 8;
    afr[ks].h[0] = cvt8_fma(Prow + base, Hrow + base, dv2);
    afr[ks].h[1] = cvt8_fma(Prow + base + 16, Hrow + base + 16, dv2);
  }

#pragma unroll
  for (int nt = 0; nt < 8; ++nt) {
    int col = nt * 16 + mn;
    v8f acc = {};                       // SRC2 = inline 0 on first WMMA
    const _Float16* wcol = wT + col * D_MODEL;
#pragma unroll
    for (int ks = 0; ks < 4; ++ks) {
      V16H bfr;
      int base = ks * 32 + half * 8;
      bfr.h[0] = *(const v8h*)(wcol + base);
      bfr.h[1] = *(const v8h*)(wcol + base + 16);
      acc = wmma_f16(afr[ks].v, bfr.v, acc);
    }
    float bias = b2[col];
#pragma unroll
    for (int r = 0; r < 8; ++r) {
      int row = r + 8 * half;  // C/D layout: lanes 0-15 -> M=r, lanes 16-31 -> M=8+r
      h2[(tile * 16 + row) * D_MODEL + col] = fmaxf(acc[r] + bias, 0.0f);
    }
  }
}

// ---------- edge MLP (WMMA): relu([h2[src]|h2[dst]] @ Wlin1 + blin1) @ Wfin + bfin -> log_softmax
// Wave owns 16 edges: A-fragments (K=256 -> 8 steps) in regs, reused across 8 N-tiles.
// Wlin1^T (f16) DMA'd into 64 KB LDS. Head (128->2) folded in-register + shfl_xor reduction.
__global__ void __launch_bounds__(256) k_edge_mlp(
    const int* __restrict__ src, const int* __restrict__ dst,
    const float* __restrict__ h2, const _Float16* __restrict__ WlinT,
    const float* __restrict__ blin1, const float* __restrict__ Wfin,
    const float* __restrict__ bfin, float* __restrict__ out) {
  __shared__ _Float16 wT[D_MODEL * 2 * D_MODEL];  // Wlin1^T: [col][k], 64 KB
  int tid = threadIdx.x;
  {
    unsigned int lds_base = (unsigned int)(uintptr_t)(void*)wT;
#pragma unroll
    for (int i = 0; i < 16; ++i) {          // 4096 x 16B chunks / 256 threads
      int idx = (tid + i * 256) * 8;        // in halves
      async_b128(lds_base + (unsigned int)idx * 2u, WlinT + idx);
    }
    wait_async0();
  }
  __syncthreads();

  int wave = tid >> 5, lane = tid & 31;
  int tile = blockIdx.x * 8 + wave;  // 50000 tiles = 6250 blocks * 8 waves, exact
  int half = lane >> 4, mn = lane & 15;
  int e = tile * 16 + mn;
  int s = src[e], d = dst[e];
  __builtin_prefetch(h2 + s * D_MODEL, 0, 3);
  __builtin_prefetch(h2 + d * D_MODEL, 0, 3);

  V16H afr[8];  // K = 256: k-steps 0-3 from h2[src], 4-7 from h2[dst]
#pragma unroll
  for (int ks = 0; ks < 8; ++ks) {
    const float* row = (ks < 4) ? (h2 + s * D_MODEL + ks * 32)
                                : (h2 + d * D_MODEL + (ks - 4) * 32);
    int base = half * 8;
    afr[ks].h[0] = cvt8(row + base);
    afr[ks].h[1] = cvt8(row + base + 16);
  }

  float wf0[8], wf1[8];
#pragma unroll
  for (int nt = 0; nt < 8; ++nt) {
    int col = nt * 16 + mn;
    wf0[nt] = Wfin[col * 2 + 0];
    wf1[nt] = Wfin[col * 2 + 1];
  }

  float pl0[8] = {}, pl1[8] = {};  // partial logits, row r = C-layout row
#pragma unroll
  for (int nt = 0; nt < 8; ++nt) {
    int col = nt * 16 + mn;
    v8f acc = {};                       // SRC2 = inline 0 on first WMMA
    const _Float16* wcol = wT + col * 256;
#pragma unroll
    for (int ks = 0; ks < 8; ++ks) {
      V16H bfr;
      int base = ks * 32 + half * 8;
      bfr.h[0] = *(const v8h*)(wcol + base);
      bfr.h[1] = *(const v8h*)(wcol + base + 16);
      acc = wmma_f16(afr[ks].v, bfr.v, acc);
    }
    float bias = blin1[col];
#pragma unroll
    for (int r = 0; r < 8; ++r) {
      float z = fmaxf(acc[r] + bias, 0.0f);  // bias + relu in epilogue
      pl0[r] += z * wf0[nt];
      pl1[r] += z * wf1[nt];
    }
  }

  // reduce over the 16 N-lanes within each half (masks 1..8 stay inside the group)
#pragma unroll
  for (int r = 0; r < 8; ++r) {
#pragma unroll
    for (int mask = 8; mask >= 1; mask >>= 1) {
      pl0[r] += __shfl_xor(pl0[r], mask, 32);
      pl1[r] += __shfl_xor(pl1[r], mask, 32);
    }
  }

  if (mn < 8) {  // lanes 0-7 write rows 0-7; lanes 16-23 write rows 8-15
    float l0 = pl0[mn] + bfin[0];
    float l1 = pl1[mn] + bfin[1];
    float mx = fmaxf(l0, l1);
    float lse = mx + logf(__expf(l0 - mx) + __expf(l1 - mx));
    int erow = tile * 16 + 8 * half + mn;
    out[erow * 2 + 0] = l0 - lse;
    out[erow * 2 + 1] = l1 - lse;
  }
}

extern "C" void kernel_launch(void* const* d_in, const int* in_sizes, int n_in,
                              void* d_out, int out_size, void* d_ws, size_t ws_size,
                              hipStream_t stream) {
  const float* x     = (const float*)d_in[0];
  const int*   ei    = (const int*)d_in[1];
  const int*   src   = ei;
  const int*   dst   = ei + N_EDGES;
  const float* W1    = (const float*)d_in[2];
  const float* b1    = (const float*)d_in[3];
  const float* W2    = (const float*)d_in[4];
  const float* b2    = (const float*)d_in[5];
  const float* Wlin1 = (const float*)d_in[6];
  const float* blin1 = (const float*)d_in[7];
  const float* Wfin  = (const float*)d_in[8];
  const float* bfin  = (const float*)d_in[9];
  float* out = (float*)d_out;

  // workspace layout (floats)
  float* ws   = (float*)d_ws;
  float* deg  = ws;              // 50000  (becomes dinv in place)
  float* xagg = ws + 50048;      // 400000
  float* h1   = ws + 450048;     // 6.4M
  float* P    = ws + 6850048;    // 6.4M
  float* h2   = ws + 13250048;   // 6.4M
  _Float16* W2T   = (_Float16*)(ws + 19650048);  // 16384 halves (32 KB)
  _Float16* WlinT = (_Float16*)(ws + 19658240);  // 32768 halves (64 KB)

  hipMemsetAsync(deg,  0, (size_t)N_NODES * sizeof(float), stream);
  hipMemsetAsync(xagg, 0, (size_t)N_NODES * D_IN * sizeof(float), stream);
  hipMemsetAsync(P,    0, (size_t)N_NODES * D_MODEL * sizeof(float), stream);

  k_cvtW  <<<(2 * D_MODEL * D_MODEL + 255) / 256, 256, 0, stream>>>(W2, Wlin1, W2T, WlinT);
  k_deg   <<<(N_EDGES + 255) / 256, 256, 0, stream>>>(dst, deg);
  k_dinv  <<<(N_NODES + 255) / 256, 256, 0, stream>>>(deg);
  k_xagg  <<<(N_EDGES + 255) / 256, 256, 0, stream>>>(src, dst, deg, x, xagg);
  k_layer1<<<(N_NODES * D_MODEL) / 256, 256, 0, stream>>>(x, xagg, deg, W1, b1, h1);
  k_hagg  <<<(N_EDGES * 32) / 256, 256, 0, stream>>>(src, dst, deg, h1, P);
  k_node_gemm<<<(N_NODES / 16 + 7) / 8, 256, 0, stream>>>(P, h1, deg, W2T, b2, h2);
  k_edge_mlp <<<(N_EDGES / 16) / 8, 256, 0, stream>>>(src, dst, h2, WlinT, blin1,
                                                      Wfin, bfin, out);
}
